// RNN_77103252897829
// MI455X (gfx1250) — matrix operand order repaired and checked by reading, not compiled, not run on hardware
//
#include <hip/hip_runtime.h>
#include <hip/hip_bf16.h>

// ---------------------------------------------------------------------------
// Elman RNN scan fused with input projection, bf16 WMMA on gfx1250 (CDNA5).
//   h_t = tanh(X_fin[:,t,:] @ W_ih^T + (b_ih+b_hh) + h_{t-1} @ W_hh^T)
// then z = relu([h_T | X_ctx] @ W_ctx^T + b_ctx);  out = z @ W_out^T + b_out
// ---------------------------------------------------------------------------

typedef __attribute__((ext_vector_type(16))) __bf16 v16bf;
typedef __attribute__((ext_vector_type(8)))  float  v8f;

#define D_IN  256
#define Hdim  512
#define D_CTX 64
#define D_OUT 128
#define Bsz   256
#define Tsz   512
#define H2    1152
#define KCAT  576

// padded LDS row strides (bf16 elements): row-to-row shift of 4 banks so the
// 16 A-rows of a b128 fragment load hit 64 distinct banks.
#define XLD 264   // 256 + 8
#define HLD 520   // 512 + 8

union Frag {
    v16bf v;
    int4  i[2];
};

__device__ __forceinline__ v8f wmma_bf16(const Frag& a, const Frag& b, v8f c) {
    // D = A(16x32 bf16) * B(32x16 bf16) + C(16x16 f32)
    return __builtin_amdgcn_wmma_f32_16x16x32_bf16(
        /*neg_a=*/false, a.v, /*neg_b=*/false, b.v,
        /*c_mod=*/(short)0, c, /*reuse_a=*/false, /*reuse_b=*/false);
}

// ---------------------------------------------------------------------------
// prep: fp32 -> bf16 weight conversion + fused bias
// ---------------------------------------------------------------------------
__global__ void prep_kernel(const float* __restrict__ Wih,
                            const float* __restrict__ Whh,
                            const float* __restrict__ bih,
                            const float* __restrict__ bhh,
                            __bf16* __restrict__ wih_b,
                            __bf16* __restrict__ whh_b,
                            float*  __restrict__ bias)
{
    int i = blockIdx.x * blockDim.x + threadIdx.x;
    if (i < Hdim * D_IN) wih_b[i] = (__bf16)Wih[i];
    if (i < Hdim * Hdim) whh_b[i] = (__bf16)Whh[i];
    if (i < Hdim)        bias[i]  = bih[i] + bhh[i];
}

// ---------------------------------------------------------------------------
// Persistent recurrent scan: 16 workgroups, one 16-row batch tile each.
// 8 waves/WG; each wave owns 4 N-tiles (4*16 = 64 of 512 output columns).
// h (16x512 bf16) lives in LDS across all T steps. Weights stream from L2
// every step. A laundered zero *offset* (not pointer!) is added to the weight
// addresses each iteration: it blocks LICM from hoisting the t-invariant
// fragment loads (which spilled to scratch), while keeping the address chain
// rooted at the kernarg pointer so the loads stay global_load_b128 (LOADcnt
// only) rather than flat_load_b128 (LOADcnt+DScnt, LDS path).
// ---------------------------------------------------------------------------
__global__ void __launch_bounds__(256)
rnn_scan_kernel(const float* __restrict__ Xfin,
                const __bf16* __restrict__ Wih,
                const __bf16* __restrict__ Whh,
                const float* __restrict__ bias,
                float* __restrict__ hT)
{
    __shared__ __bf16 h_lds[16 * HLD];   // 16.25 KB: hidden state, bf16
    __shared__ __bf16 x_lds[16 * XLD];   //  8.25 KB: staged x_t tile, bf16

    const int tid  = threadIdx.x;
    const int wave = tid >> 5;
    const int lane = tid & 31;
    const int l16  = lane & 15;      // A: row m     | B: col n     | C: col n
    const int hl   = lane >> 4;      // half-wave select
    const int kbaseA = hl * 8;       // A-frag lane k-chunk base
    const int koffB  = hl * 8;       // B-frag lane k-chunk base
    const int b0 = blockIdx.x * 16;  // batch tile base row

    // zero the hidden state (h_{-1} = 0)
    for (int i = tid; i < 16 * HLD; i += 256) h_lds[i] = (__bf16)0.0f;

    // preload per-column bias (b_ih + b_hh); C layout: lane column = l16
    float bv[4];
#pragma unroll
    for (int nt = 0; nt < 4; ++nt) {
        const int n0 = (wave * 4 + nt) * 16;
        bv[nt] = bias[n0 + l16];
    }

    // per-lane element offsets of this lane's B fragment within a weight row
    // block; the n-dependent part is folded in here, the k part comes from kt.
    size_t offW[4];   // offset into Wih (elements)
    size_t offH[4];   // offset into Whh (elements)
#pragma unroll
    for (int nt = 0; nt < 4; ++nt) {
        const int n = (wave * 4 + nt) * 16 + l16;
        offW[nt] = (size_t)n * D_IN + koffB;
        offH[nt] = (size_t)n * Hdim + koffB;
    }

    // x staging assignment: thread -> (row, 4 float4 chunks), coalesced
    const int xrow  = tid >> 4;      // 0..15
    const int xlane = tid & 15;      // 0..15

    __syncthreads();

    for (int t = 0; t < Tsz; ++t) {
        // Laundered zero offset: redefined by volatile asm every iteration so
        // the weight fragment loads below are not loop-invariant and cannot
        // be hoisted out of the t-loop. Lives in an SGPR pair; addresses stay
        // rooted at the (global) kernarg pointers.
        size_t lz = 0;
        asm volatile("" : "+s"(lz));
        const __bf16* WihT = Wih + lz;
        const __bf16* WhhT = Whh + lz;

        // ---- stage x_t tile: 16 x 256 f32 -> bf16 into LDS ----
        const float* xsrc = Xfin + ((size_t)(b0 + xrow) * Tsz + t) * D_IN;
#pragma unroll
        for (int c = xlane; c < 64; c += 16) {
            float4 f = *(const float4*)(xsrc + c * 4);
            union { __bf16 b[4]; int2 p; } u;
            u.b[0] = (__bf16)f.x; u.b[1] = (__bf16)f.y;
            u.b[2] = (__bf16)f.z; u.b[3] = (__bf16)f.w;
            *(int2*)(x_lds + xrow * XLD + c * 4) = u.p;
        }
        __syncthreads();   // x ready; h from previous step ready

        // ---- fused GEMM: acc = x_t @ Wih^T + h @ Whh^T ----
        // kt outer / nt inner: one A fragment (2x ds_load_b128) feeds 4 WMMAs.
        v8f acc[4] = {{}, {}, {}, {}};

        // K = 256 over staged x (A from LDS, B = Wih rows from L2)
        {
            const __bf16* arow = x_lds + l16 * XLD + kbaseA;
#pragma unroll
            for (int kt = 0; kt < 8; ++kt) {
                Frag a;
                a.i[0] = *(const int4*)(arow + kt * 32);
                a.i[1] = *(const int4*)(arow + kt * 32 + 16);
#pragma unroll
                for (int nt = 0; nt < 4; ++nt) {
                    Frag b;
                    const __bf16* bp = WihT + offW[nt] + kt * 32;
                    b.i[0] = *(const int4*)(bp);
                    b.i[1] = *(const int4*)(bp + 16);
                    acc[nt] = wmma_bf16(a, b, acc[nt]);
                }
            }
        }

        // K = 512 over hidden state (A from LDS, B = Whh rows from L2)
        {
            const __bf16* arow = h_lds + l16 * HLD + kbaseA;
#pragma unroll
            for (int kt = 0; kt < 16; ++kt) {
                Frag a;
                a.i[0] = *(const int4*)(arow + kt * 32);
                a.i[1] = *(const int4*)(arow + kt * 32 + 16);
#pragma unroll
                for (int nt = 0; nt < 4; ++nt) {
                    Frag b;
                    const __bf16* bp = WhhT + offH[nt] + kt * 32;
                    b.i[0] = *(const int4*)(bp);
                    b.i[1] = *(const int4*)(bp + 16);
                    acc[nt] = wmma_bf16(a, b, acc[nt]);
                }
            }
        }
        __syncthreads();   // all reads of h_lds / x_lds complete

        // ---- bias + tanh, write new h back to LDS (bf16) ----
#pragma unroll
        for (int nt = 0; nt < 4; ++nt) {
            const int n = (wave * 4 + nt) * 16 + l16;
#pragma unroll
            for (int r = 0; r < 8; ++r) {
                const int m = r + 8 * hl;          // C layout row
                float y = tanhf(acc[nt][r] + bv[nt]);
                h_lds[m * HLD + n] = (__bf16)y;
                if (t == Tsz - 1)
                    hT[(size_t)(b0 + m) * Hdim + n] = y;  // full-precision h_T
            }
        }
        // next iteration's x-staging + its barrier orders these h writes
        // before any reads at step t+1.
    }
}

// ---------------------------------------------------------------------------
// z = relu([h_T | X_ctx] @ W_ctx^T + b_ctx)   (256 x 1152, tiny: fp32 VALU)
// ---------------------------------------------------------------------------
__global__ void ctx_kernel(const float* __restrict__ hT,
                           const float* __restrict__ Xctx,
                           const float* __restrict__ Wctx,
                           const float* __restrict__ bctx,
                           float* __restrict__ z)
{
    int j = blockIdx.x * blockDim.x + threadIdx.x;   // 0..1151
    int m = blockIdx.y;                              // 0..255
    if (j >= H2) return;
    const float* w  = Wctx + (size_t)j * KCAT;
    const float* hr = hT   + (size_t)m * Hdim;
    const float* xc = Xctx + (size_t)m * D_CTX;
    float s = bctx[j];
    for (int k = 0; k < Hdim; ++k)  s += hr[k] * w[k];
    for (int k = 0; k < D_CTX; ++k) s += xc[k] * w[Hdim + k];
    z[(size_t)m * H2 + j] = fmaxf(s, 0.0f);
}

// ---------------------------------------------------------------------------
// out = z @ W_out^T + b_out   (256 x 128, tiny: fp32 VALU)
// ---------------------------------------------------------------------------
__global__ void out_kernel(const float* __restrict__ z,
                           const float* __restrict__ Wout,
                           const float* __restrict__ bout,
                           float* __restrict__ out)
{
    int o = threadIdx.x;   // 0..127
    int m = blockIdx.x;    // 0..255
    const float* w  = Wout + (size_t)o * H2;
    const float* zr = z    + (size_t)m * H2;
    float s = bout[o];
    for (int k = 0; k < H2; ++k) s += zr[k] * w[k];
    out[(size_t)m * D_OUT + o] = s;
}

// ---------------------------------------------------------------------------
extern "C" void kernel_launch(void* const* d_in, const int* in_sizes, int n_in,
                              void* d_out, int out_size, void* d_ws, size_t ws_size,
                              hipStream_t stream)
{
    const float* Xfin = (const float*)d_in[0];
    const float* Xctx = (const float*)d_in[1];
    const float* Wih  = (const float*)d_in[2];
    const float* Whh  = (const float*)d_in[3];
    const float* bih  = (const float*)d_in[4];
    const float* bhh  = (const float*)d_in[5];
    const float* Wctx = (const float*)d_in[6];
    const float* bctx = (const float*)d_in[7];
    const float* Wout = (const float*)d_in[8];
    const float* bout = (const float*)d_in[9];
    float* out = (float*)d_out;

    // workspace layout (~2.5 MB total)
    char* ws = (char*)d_ws;
    __bf16* wih_b = (__bf16*)(ws);                                  // 256 KB
    __bf16* whh_b = (__bf16*)(ws + 262144);                         // 512 KB
    float*  bias  = (float*) (ws + 262144 + 524288);                //   2 KB
    float*  hT    = (float*) (ws + 262144 + 524288 + 2048);         // 512 KB
    float*  z     = (float*) (ws + 262144 + 524288 + 2048 + 524288);// 1.125 MB

    prep_kernel<<<(Hdim * Hdim + 255) / 256, 256, 0, stream>>>(
        Wih, Whh, bih, bhh, wih_b, whh_b, bias);

    rnn_scan_kernel<<<Bsz / 16, 256, 0, stream>>>(Xfin, wih_b, whh_b, bias, hT);

    ctx_kernel<<<dim3(H2 / 128, Bsz), 128, 0, stream>>>(hT, Xctx, Wctx, bctx, z);

    out_kernel<<<Bsz, D_OUT, 0, stream>>>(z, Wout, bout, out);
}
